// GatedMultiHeadSelfAttention_43628277792943
// MI455X (gfx1250) — matrix operand order repaired and checked by reading, not compiled, not run on hardware
//
#include <hip/hip_runtime.h>
#include <hip/hip_bf16.h>

typedef __attribute__((ext_vector_type(16))) _Float16 v16h;
typedef __attribute__((ext_vector_type(8)))  float    v8f;

#define EMBED  768
#define NHEAD  12
#define HD     64
#define BB     4
#define TT     2048
#define BT     (BB*TT)        // 8192
#define QKVN   (3*EMBED)      // 2304

// ---------------------------------------------------------------------------
// Tensor Data Mover support (guarded; cooperative-copy fallback otherwise)
// ---------------------------------------------------------------------------
#if defined(__gfx1250__) && __has_builtin(__builtin_amdgcn_tensor_load_to_lds) && \
    __has_builtin(__builtin_amdgcn_s_wait_tensorcnt)
#define USE_TDM 1
#else
#define USE_TDM 0
#endif

#if USE_TDM
typedef __attribute__((ext_vector_type(4))) unsigned int u32x4;
typedef __attribute__((ext_vector_type(8))) int i32x8;
typedef __attribute__((ext_vector_type(4))) int i32x4;

// Low 32 bits of a generic pointer to __shared__ == LDS byte offset
__device__ __forceinline__ unsigned lds_off_of(const void* p) {
  return (unsigned)(uintptr_t)p;
}

// Build a D# (ISA 8.3-8.6) for a [tile_d x tile_h x tile_w] fp16 tile and issue
// TENSOR_LOAD_TO_LDS. stride0 = elements between rows, stride1 = elements
// between z-slices. tile_d==0 -> plain 2D tile.
__device__ __forceinline__ void tdm_load_tile(unsigned lds_byte, const void* gsrc,
                                              unsigned tile_w, unsigned tile_h,
                                              unsigned tile_d,
                                              unsigned tdim0, unsigned tdim1,
                                              unsigned stride0, unsigned stride1,
                                              unsigned tdim2) {
  unsigned long long ga = (unsigned long long)(uintptr_t)gsrc;
  u32x4 g0;
  g0[0] = 1u;                                             // count=1, user desc
  g0[1] = lds_byte;                                       // lds_addr
  g0[2] = (unsigned)ga;                                   // global_addr[31:0]
  g0[3] = ((unsigned)(ga >> 32) & 0x1FFFFFFu) | (2u << 30); // addr[56:32] | type=2
  i32x8 g1;
  g1[0] = (int)(1u << 16);                                // data_size = 2 bytes
  g1[1] = (int)((tdim0 & 0xFFFFu) << 16);                 // tensor_dim0[15:0]
  g1[2] = (int)((tdim0 >> 16) | ((tdim1 & 0xFFFFu) << 16));
  g1[3] = (int)((tdim1 >> 16) | (tile_w << 16));          // tile_dim0
  g1[4] = (int)(tile_h | (tile_d << 16));                 // tile_dim1 | tile_dim2
  g1[5] = (int)stride0;                                   // tensor_dim0_stride[31:0]
  g1[6] = (int)((stride1 & 0xFFFFu) << 16);               // s0 hi16=0 | s1[15:0]
  g1[7] = (int)(stride1 >> 16);                           // s1[47:16]
  i32x4 g2; g2[0] = (int)tdim2; g2[1] = 0; g2[2] = 0; g2[3] = 0;
  i32x4 g3 = {0, 0, 0, 0};
#if __clang_major__ >= 23
  i32x8 g4 = {0, 0, 0, 0, 0, 0, 0, 0};
  __builtin_amdgcn_tensor_load_to_lds(g0, g1, g2, g3, g4, 0);
#else
  __builtin_amdgcn_tensor_load_to_lds(g0, g1, g2, g3, 0);
#endif
}
#endif  // USE_TDM

// ---------------------------------------------------------------------------
// WMMA fragment loaders (wave32; contiguous k order, element s of lane half h holds k = 16h + s)
// ---------------------------------------------------------------------------

// A fragment: 16x32 f16, row-major source, p -> (row0,k0)
__device__ __forceinline__ v16h load_a_frag(const _Float16* p, int ld) {
  int lane = threadIdx.x & 31;
  int m = lane & 15, lh = lane >> 4;
  const _Float16* row = p + (size_t)m * ld;
  v16h a;
#pragma unroll
  for (int r = 0; r < 8; ++r) {
    int kb = lh * 16 + 2 * r;
    a[2 * r]     = row[kb];
    a[2 * r + 1] = row[kb + 1];
  }
  return a;
}

// B fragment: 32x16 f16 (KxN), row-major source, p -> (k0,col0)
__device__ __forceinline__ v16h load_b_frag(const _Float16* p, int ld) {
  int lane = threadIdx.x & 31;
  int n = lane & 15, lh = lane >> 4;
  const _Float16* col = p + n + (size_t)(lh * 16) * ld;
  v16h b;
#pragma unroll
  for (int r = 0; r < 8; ++r) {
    b[2 * r]     = col[(size_t)(2 * r) * ld];
    b[2 * r + 1] = col[(size_t)(2 * r + 1) * ld];
  }
  return b;
}

// B fragment of transposed source: B[kk][n] = p[n*ld + kk] (for Q.K^T) —
// contiguous 32B per lane.
__device__ __forceinline__ v16h load_bT_frag(const _Float16* p, int ld) {
  int lane = threadIdx.x & 31;
  int n = lane & 15, lh = lane >> 4;
  const _Float16* row = p + (size_t)n * ld + lh * 16;
  v16h b;
#pragma unroll
  for (int r = 0; r < 8; ++r) {
    b[2 * r]     = row[2 * r];
    b[2 * r + 1] = row[2 * r + 1];
  }
  return b;
}

// ---------------------------------------------------------------------------
// Prep kernels
// ---------------------------------------------------------------------------

__global__ void prep_x(const float* __restrict__ hs, _Float16* __restrict__ X16, int n) {
  int i = blockIdx.x * blockDim.x + threadIdx.x;
  if (i < n) X16[i] = (_Float16)hs[i];
}

__global__ void prep_wqkv(const float* __restrict__ Wq, const float* __restrict__ Wk,
                          const float* __restrict__ Wv, _Float16* __restrict__ W16) {
  int i = blockIdx.x * blockDim.x + threadIdx.x;   // over H*D*hd = 589824
  if (i >= NHEAD * EMBED * HD) return;
  int e = i % HD;
  int d = (i / HD) % EMBED;
  int h = i / (HD * EMBED);
  int col = h * HD + e;
  W16[(size_t)d * QKVN + col]             = (_Float16)Wq[i];
  W16[(size_t)d * QKVN + EMBED + col]     = (_Float16)Wk[i];
  W16[(size_t)d * QKVN + 2 * EMBED + col] = (_Float16)Wv[i];
}

__global__ void prep_bqkv(const float* __restrict__ bq, const float* __restrict__ bk,
                          const float* __restrict__ bv, float* __restrict__ bias) {
  int i = blockIdx.x * blockDim.x + threadIdx.x;   // 768 (flat h*64+e == col)
  if (i >= EMBED) return;
  bias[i]             = bq[i];
  bias[EMBED + i]     = bk[i];
  bias[2 * EMBED + i] = bv[i];
}

__global__ void prep_wo(const float* __restrict__ Wo, const float* __restrict__ gate,
                        _Float16* __restrict__ Wg) {
  int i = blockIdx.x * blockDim.x + threadIdx.x;   // (h*64+e)*768 + d
  if (i >= NHEAD * HD * EMBED) return;
  int h = i / (HD * EMBED);
  float g = gate[h];
  g = (g < 1e-6f) ? 0.0f : g;
  Wg[i] = (_Float16)(g * Wo[i]);
}

__global__ void prep_bo(const float* __restrict__ bo, const float* __restrict__ gate,
                        float* __restrict__ bsum) {
  int d = blockIdx.x * blockDim.x + threadIdx.x;
  if (d >= EMBED) return;
  float s = 0.0f;
#pragma unroll
  for (int h = 0; h < NHEAD; ++h) {
    float g = gate[h];
    g = (g < 1e-6f) ? 0.0f : g;
    s += g * bo[h * EMBED + d];
  }
  bsum[d] = s;
}

// ---------------------------------------------------------------------------
// WMMA GEMM: C[M,N] = A[M,K](f16) * B[K,N](f16) + bias[N]
// Block = 8 waves = 128x64 tile; shared 32x64 B tile double-buffered in LDS,
// DMA'd by the Tensor Data Mover (wave 0) and overlapped with compute.
// ---------------------------------------------------------------------------

template <bool F16OUT>
__global__ void gemm_wmma(const _Float16* __restrict__ A, const _Float16* __restrict__ Bm,
                          const float* __restrict__ bias, void* __restrict__ Cout,
                          int N, int K, int ntB) {
  __shared__ __align__(16) _Float16 Bt[2][32 * 64];
  int nt = blockIdx.x % ntB;
  int mb = blockIdx.x / ntB;
  int warp = threadIdx.x >> 5;
  int m0 = mb * 128 + warp * 16;
  int n0 = nt * 64;
  int lane = threadIdx.x & 31;
  int n = lane & 15, lh = lane >> 4;

  v8f acc[4];
#pragma unroll
  for (int j = 0; j < 4; ++j)
#pragma unroll
    for (int r = 0; r < 8; ++r) acc[j][r] = 0.0f;

  const int niter = K / 32;
#if USE_TDM
  if (threadIdx.x < 32)
    tdm_load_tile(lds_off_of(&Bt[0][0]), Bm + n0, 64, 32, 0,
                  (unsigned)N, (unsigned)K, (unsigned)N, 0, 0);
#endif
  for (int i = 0; i < niter; ++i) {
    int k0 = i * 32;
#if USE_TDM
    if (threadIdx.x < 32) {
      if (i + 1 < niter) {
        tdm_load_tile(lds_off_of(&Bt[(i + 1) & 1][0]),
                      Bm + (size_t)(k0 + 32) * N + n0, 64, 32, 0,
                      (unsigned)N, (unsigned)K, (unsigned)N, 0, 0);
        __builtin_amdgcn_s_wait_tensorcnt(1);   // oldest (tile i) complete
      } else {
        __builtin_amdgcn_s_wait_tensorcnt(0);   // last tile complete
      }
    }
#else
    {
      int t = threadIdx.x;                 // 256 threads x one uint4 = 32x64 tile
      int c0 = (t & 7) * 8, row = t >> 3;
      *(uint4*)&Bt[i & 1][row * 64 + c0] =
          *(const uint4*)(Bm + (size_t)(k0 + row) * N + n0 + c0);
    }
#endif
    __syncthreads();
    v16h a = load_a_frag(A + (size_t)m0 * K + k0, K);
    const _Float16* Bl = &Bt[i & 1][0];
#pragma unroll
    for (int j = 0; j < 4; ++j) {
      v16h b = load_b_frag(Bl + 16 * j, 64);
      acc[j] = __builtin_amdgcn_wmma_f32_16x16x32_f16(
          false, a, false, b, (short)0, acc[j], false, false);
    }
    __syncthreads();
  }

#pragma unroll
  for (int j = 0; j < 4; ++j) {
#pragma unroll
    for (int r = 0; r < 8; ++r) {
      int col = n0 + 16 * j + n;
      float v = acc[j][r] + bias[col];
      size_t idx = (size_t)(m0 + r + 8 * lh) * N + col;
      if constexpr (F16OUT) ((_Float16*)Cout)[idx] = (_Float16)v;
      else                  ((float*)Cout)[idx] = v;
    }
  }
}

// ---------------------------------------------------------------------------
// Fused flash attention. Block = one (b,h) x 8 query tiles (8 waves).
// Per 32-key step, one 3D TDM descriptor (64 x 32 x 2, z-stride 768) pulls the
// K tile and V tile into LDS (double-buffered), shared by all 8 waves.
// QKV layout: [BT][2304] fp16, cols: Q=h*64+e, K=768+h*64+e, V=1536+h*64+e.
// ---------------------------------------------------------------------------

__global__ void attn_kernel(const _Float16* __restrict__ QKV, _Float16* __restrict__ AO) {
  __shared__ __align__(16) _Float16 KV[2][2][32 * 64];  // [buf][K/V][row*64+col]
  __shared__ __align__(16) _Float16 Pbuf[8][16 * 32];
  int warp = threadIdx.x >> 5;
  int lane = threadIdx.x & 31;
  int n = lane & 15, lh = lane >> 4;

  int blk = blockIdx.x;             // 768 = B * H * 16
  int qb = blk & 15;
  int h  = (blk >> 4) % NHEAD;
  int b  = blk / (16 * NHEAD);
  int qt = qb * 8 + warp;
  size_t bt0 = (size_t)b * TT + (size_t)qt * 16;

  v16h aQ0 = load_a_frag(QKV + bt0 * QKVN + h * HD, QKVN);
  v16h aQ1 = load_a_frag(QKV + bt0 * QKVN + h * HD + 32, QKVN);

  v8f O0, O1, O2, O3;
  float rmax[8], rsum[8];
#pragma unroll
  for (int r = 0; r < 8; ++r) {
    O0[r] = 0.0f; O1[r] = 0.0f; O2[r] = 0.0f; O3[r] = 0.0f;
    rmax[r] = -1e30f; rsum[r] = 0.0f;
  }
  _Float16* Pw = Pbuf[warp];

  const int niter = TT / 32;        // 64
#if USE_TDM
  if (threadIdx.x < 32)
    tdm_load_tile(lds_off_of(&KV[0][0][0]),
                  QKV + (size_t)b * TT * QKVN + EMBED + h * HD,
                  64, 32, 2, QKVN, BT, QKVN, EMBED, 2);
#endif
  for (int i = 0; i < niter; ++i) {
    int kt = i * 32;
#if USE_TDM
    if (threadIdx.x < 32) {
      if (i + 1 < niter) {
        tdm_load_tile(lds_off_of(&KV[(i + 1) & 1][0][0]),
                      QKV + ((size_t)b * TT + kt + 32) * QKVN + EMBED + h * HD,
                      64, 32, 2, QKVN, BT, QKVN, EMBED, 2);
        __builtin_amdgcn_s_wait_tensorcnt(1);   // oldest (tile i) complete
      } else {
        __builtin_amdgcn_s_wait_tensorcnt(0);   // last tile complete
      }
    }
#else
    {
      const _Float16* src = QKV + ((size_t)b * TT + kt) * QKVN + EMBED + h * HD;
      int t = threadIdx.x;           // 256 threads x two uint4 = 2 x 32 x 64 tile
#pragma unroll
      for (int q = 0; q < 2; ++q) {
        int f8 = t + q * 256;
        int c0 = (f8 & 7) * 8;
        int row = (f8 >> 3) & 31;
        int z = f8 >> 8;             // 0=K, 1=V
        *(uint4*)&KV[i & 1][z][row * 64 + c0] =
            *(const uint4*)(src + (size_t)row * QKVN + (size_t)z * EMBED + c0);
      }
    }
#endif
    __syncthreads();                                  // (a) tile i ready
    const _Float16* Klds = &KV[i & 1][0][0];
    const _Float16* Vlds = &KV[i & 1][1][0];

    v8f S0, S1;
#pragma unroll
    for (int r = 0; r < 8; ++r) { S0[r] = 0.0f; S1[r] = 0.0f; }
    S0 = __builtin_amdgcn_wmma_f32_16x16x32_f16(false, aQ0, false, load_bT_frag(Klds, 64),            (short)0, S0, false, false);
    S0 = __builtin_amdgcn_wmma_f32_16x16x32_f16(false, aQ1, false, load_bT_frag(Klds + 32, 64),       (short)0, S0, false, false);
    S1 = __builtin_amdgcn_wmma_f32_16x16x32_f16(false, aQ0, false, load_bT_frag(Klds + 16 * 64, 64),      (short)0, S1, false, false);
    S1 = __builtin_amdgcn_wmma_f32_16x16x32_f16(false, aQ1, false, load_bT_frag(Klds + 16 * 64 + 32, 64), (short)0, S1, false, false);

    // Online softmax over the 32 new keys (rows live on 16-lane halves).
#pragma unroll
    for (int r = 0; r < 8; ++r) {
      float s0 = S0[r] * 0.125f;     // 1/sqrt(64)
      float s1 = S1[r] * 0.125f;
      float mx = fmaxf(s0, s1);
      mx = fmaxf(mx, __shfl_xor(mx, 1));
      mx = fmaxf(mx, __shfl_xor(mx, 2));
      mx = fmaxf(mx, __shfl_xor(mx, 4));
      mx = fmaxf(mx, __shfl_xor(mx, 8));
      float nm = fmaxf(rmax[r], mx);
      float corr = __expf(rmax[r] - nm);
      float p0 = __expf(s0 - nm);
      float p1 = __expf(s1 - nm);
      float ps = p0 + p1;
      ps += __shfl_xor(ps, 1);
      ps += __shfl_xor(ps, 2);
      ps += __shfl_xor(ps, 4);
      ps += __shfl_xor(ps, 8);
      rsum[r] = rsum[r] * corr + ps;
      rmax[r] = nm;
      O0[r] *= corr; O1[r] *= corr; O2[r] *= corr; O3[r] *= corr;
      Pw[(r + 8 * lh) * 32 + n]      = (_Float16)p0;
      Pw[(r + 8 * lh) * 32 + 16 + n] = (_Float16)p1;
    }
    __syncthreads();                                  // (b) P staged

    v16h aP = load_a_frag(Pw, 32);
    O0 = __builtin_amdgcn_wmma_f32_16x16x32_f16(false, aP, false, load_b_frag(Vlds,      64), (short)0, O0, false, false);
    O1 = __builtin_amdgcn_wmma_f32_16x16x32_f16(false, aP, false, load_b_frag(Vlds + 16, 64), (short)0, O1, false, false);
    O2 = __builtin_amdgcn_wmma_f32_16x16x32_f16(false, aP, false, load_b_frag(Vlds + 32, 64), (short)0, O2, false, false);
    O3 = __builtin_amdgcn_wmma_f32_16x16x32_f16(false, aP, false, load_b_frag(Vlds + 48, 64), (short)0, O3, false, false);
    __syncthreads();                                  // (c) buf i free for reuse
  }

#pragma unroll
  for (int r = 0; r < 8; ++r) {
    float is = 1.0f / rsum[r];
    size_t base = (bt0 + r + 8 * lh) * EMBED + h * HD + n;
    AO[base]      = (_Float16)(O0[r] * is);
    AO[base + 16] = (_Float16)(O1[r] * is);
    AO[base + 32] = (_Float16)(O2[r] * is);
    AO[base + 48] = (_Float16)(O3[r] * is);
  }
}

// ---------------------------------------------------------------------------
// Launch
// ---------------------------------------------------------------------------

extern "C" void kernel_launch(void* const* d_in, const int* in_sizes, int n_in,
                              void* d_out, int out_size, void* d_ws, size_t ws_size,
                              hipStream_t stream) {
  (void)in_sizes; (void)n_in; (void)out_size; (void)ws_size;
  const float* hs   = (const float*)d_in[0];
  const float* Wq   = (const float*)d_in[1];
  const float* bq   = (const float*)d_in[2];
  const float* Wk   = (const float*)d_in[3];
  const float* bk   = (const float*)d_in[4];
  const float* Wv   = (const float*)d_in[5];
  const float* bv   = (const float*)d_in[6];
  const float* Wo   = (const float*)d_in[7];
  const float* bo   = (const float*)d_in[8];
  const float* gate = (const float*)d_in[9];

  char* ws = (char*)d_ws;
  size_t off = 0;
  _Float16* X16    = (_Float16*)(ws + off); off += (size_t)BT * EMBED * 2;
  _Float16* Wqkv16 = (_Float16*)(ws + off); off += (size_t)EMBED * QKVN * 2;
  float*    bqkv   = (float*)   (ws + off); off += (size_t)QKVN * 4;
  _Float16* Wg16   = (_Float16*)(ws + off); off += (size_t)EMBED * EMBED * 2;
  float*    bsum   = (float*)   (ws + off); off += (size_t)EMBED * 4;
  _Float16* QKV16  = (_Float16*)(ws + off); off += (size_t)BT * QKVN * 2;
  _Float16* A16    = (_Float16*)(ws + off); off += (size_t)BT * EMBED * 2;

  const int nx = BT * EMBED;                 // 6291456
  const int nw = NHEAD * EMBED * HD;         // 589824
  prep_x   <<<(nx + 255) / 256, 256, 0, stream>>>(hs, X16, nx);
  prep_wqkv<<<(nw + 255) / 256, 256, 0, stream>>>(Wq, Wk, Wv, Wqkv16);
  prep_bqkv<<<3, 256, 0, stream>>>(bq, bk, bv, bqkv);
  prep_wo  <<<(nw + 255) / 256, 256, 0, stream>>>(Wo, gate, Wg16);
  prep_bo  <<<3, 256, 0, stream>>>(bo, gate, bsum);

  // QKV GEMM: 8192x2304x768. (8192/128)*(2304/64) = 64*36 = 2304 blocks.
  gemm_wmma<true><<<2304, 256, 0, stream>>>(X16, Wqkv16, bqkv, QKV16,
                                            QKVN, EMBED, QKVN / 64);

  // Attention: B*H*16 = 768 blocks (8 q-tiles of one (b,h) per block).
  attn_kernel<<<768, 256, 0, stream>>>(QKV16, A16);

  // Output GEMM: 8192x768x768 -> fp32 d_out. 64*12 = 768 blocks.
  gemm_wmma<false><<<768, 256, 0, stream>>>(A16, Wg16, bsum, d_out,
                                            EMBED, EMBED, EMBED / 64);
}
